// PointTrans_Layer_64742337020257
// MI455X (gfx1250) — matrix operand order repaired and checked by reading, not compiled, not run on hardware
//
#include <hip/hip_runtime.h>

#define Nn 50000
#define Ee 800000
#define ET 850000   // E + N self loops
#define BN_S 0.9999950000374997f   // 1/sqrt(1+1e-5)

typedef __attribute__((ext_vector_type(16))) _Float16 v16h;
typedef __attribute__((ext_vector_type(8)))  _Float16 v8h;
typedef __attribute__((ext_vector_type(8)))  float    v8f;
typedef __attribute__((ext_vector_type(4)))  unsigned v4u;
typedef __attribute__((ext_vector_type(8)))  int      v8i;
typedef __attribute__((ext_vector_type(4)))  int      v4i;

// ---------------------------------------------------------------- helpers ---

__device__ __forceinline__ v8f wmma_f16(v16h a, v16h b, v8f c) {
    return __builtin_amdgcn_wmma_f32_16x16x32_f16(
        false, a, false, b, (short)0, c, false, false);
}

// Packed f16 weight layout (built once by pack_weights_kernel):
// per matrix: frag f = ktile*4+ntile (0..7), lane l (0..31), half h (0..15)
// at offset ((f*32 + l)*16 + h); each lane's fragment is 32B contiguous.
__device__ __forceinline__ v16h load_bfrag_lds(const _Float16* Wp, int frag, int lane) {
    const v8h* p  = (const v8h*)(Wp + ((frag * 32 + lane) << 4));
    v8h lo8 = p[0], hi8 = p[1];
    v16h b;
#pragma unroll
    for (int i = 0; i < 8; ++i) { b[i] = lo8[i]; b[i + 8] = hi8[i]; }
    return b;
}

__device__ __forceinline__ v16h make_afrag(v8h lo8, v8h hi8) {
    v16h a;
#pragma unroll
    for (int i = 0; i < 8; ++i) { a[i] = lo8[i]; a[i + 8] = hi8[i]; }
    return a;
}

// D = A[16x64] * W[64x64] with LDS-resident packed weights (8 WMMAs)
__device__ __forceinline__ void gemm64(v8f d[4], v16h a0, v16h a1,
                                       const _Float16* Wp, int lane) {
#pragma unroll
    for (int nt = 0; nt < 4; ++nt) {
        v8f c = {0.f, 0.f, 0.f, 0.f, 0.f, 0.f, 0.f, 0.f};
        c = wmma_f16(a0, load_bfrag_lds(Wp, nt,     lane), c);
        c = wmma_f16(a1, load_bfrag_lds(Wp, 4 + nt, lane), c);
        d[nt] = c;
    }
}

// relu((d + b[c]) * BN_S*g[c] + t[c]) with c = ntile*16 + (lane&15)
__device__ __forceinline__ void bias_bn_relu(v8f d[4], const float* b, const float* g,
                                             const float* t, int lo) {
#pragma unroll
    for (int nt = 0; nt < 4; ++nt) {
        int   c  = nt * 16 + lo;
        float s  = BN_S * g[c];
        float tt = b[c] * s + t[c];
#pragma unroll
        for (int v = 0; v < 8; ++v) d[nt][v] = fmaxf(d[nt][v] * s + tt, 0.f);
    }
}

struct Chunks { v8h c0, c1, c2, c3; };

// D layout (lane: col nt*16+lo, rows 8*hi+v) -> row-major f16 LDS tile,
// reload as per-lane A-fragment chunks (row lo, K chunks {0,16,32,48}+8*hi).
__device__ __forceinline__ Chunks transpose_D(_Float16* lds, const v8f d[4], int lane) {
    int lo = lane & 15, hi = lane >> 4;
#pragma unroll
    for (int nt = 0; nt < 4; ++nt)
#pragma unroll
        for (int v = 0; v < 8; ++v)
            lds[(8 * hi + v) * 64 + nt * 16 + lo] = (_Float16)d[nt][v];
    __asm__ volatile("s_wait_dscnt 0" ::: "memory");
    const v8h* row = (const v8h*)(lds + lo * 64);
    Chunks ch;
    ch.c0 = row[0 + hi]; ch.c1 = row[2 + hi];
    ch.c2 = row[4 + hi]; ch.c3 = row[6 + hi];
    return ch;
}

__device__ __forceinline__ void store_chunks(_Float16* dst, long rowbase, Chunks ch, int lane) {
    int lo = lane & 15, hi = lane >> 4;
    v8h* g = (v8h*)(dst + (rowbase + lo) * 64);
    g[0 + hi] = ch.c0; g[2 + hi] = ch.c1; g[4 + hi] = ch.c2; g[6 + hi] = ch.c3;
}

// ------------------------------------------------------------ TDM staging ---

__device__ __forceinline__ void set_bits(unsigned long long* a, int pos, int width,
                                         unsigned long long val) {
    int w = pos >> 6, b = pos & 63;
    a[w] |= (val << b);
    if (b + width > 64) a[w + 1] |= (val >> (64 - b));
}

// Stage `ndwords` contiguous dwords from global `src` into LDS offset `lds_off`
// via the Tensor Data Mover (D# per CDNA5 ISA).
// Issue from one wave, then s_wait_tensorcnt; caller barriers the block.
// This toolchain exposes the 6-arg builtin:
//   (uint32x4 g0, int32x8 g1, int32x4 g2, int32x4 g3, int32x8 g4, i32 cpol)
__device__ __forceinline__ void tdm_stage(unsigned lds_off, const void* src, unsigned ndwords) {
    unsigned long long g0b[2] = {0ull, 0ull};
    unsigned long long g1b[4] = {0ull, 0ull, 0ull, 0ull};
    unsigned long long ga = (unsigned long long)(__SIZE_TYPE__)src;
    // group0: count=1 | lds_addr[63:32] | global_addr[120:64] | type=2 [127:126]
    set_bits(g0b, 0, 2, 1ull);
    set_bits(g0b, 32, 32, (unsigned long long)lds_off);
    set_bits(g0b, 64, 57, ga & ((1ull << 57) - 1));
    set_bits(g0b, 126, 2, 2ull);
    // group1: data_size=2 (4B) [17:16]; tensor_dim0 [79:48]; tensor_dim1 [111:80];
    //         tile_dim0 [127:112]; tile_dim1 [143:128]; tensor_dim0_stride [207:160]
    set_bits(g1b, 16, 2, 2ull);
    set_bits(g1b, 48, 32, (unsigned long long)ndwords);
    set_bits(g1b, 80, 32, 1ull);
    set_bits(g1b, 112, 16, (unsigned long long)ndwords);
    set_bits(g1b, 128, 16, 1ull);
    set_bits(g1b, 160, 48, (unsigned long long)ndwords);
    v4u g0; v8i g1;
    v4i gz4 = {0, 0, 0, 0};
    v8i gz8 = {0, 0, 0, 0, 0, 0, 0, 0};
    __builtin_memcpy(&g0, g0b, 16);
    __builtin_memcpy(&g1, g1b, 32);
    __builtin_amdgcn_tensor_load_to_lds(g0, g1, gz4, gz4, gz8, 0);
    __builtin_amdgcn_s_wait_tensorcnt(0);
}

// ---------------------------------------------------------------- kernels ---

__global__ void zero_kernel(unsigned* p, long n) {
    long i = (long)blockIdx.x * blockDim.x + threadIdx.x;
    if (i < n) p[i] = 0u;
}

// Convert 7 weight matrices (64x64 f32, row-major W[k][n]) to packed f16
// B-fragment layout. matrices: 0 W_lin, 1 W_src, 2 W_dst, 3 pW2, 4 aW1,
// 5 aW2, 6 W_up; 4096 halves each.
__global__ void pack_weights_kernel(const float* __restrict__ W0, const float* __restrict__ W1,
                                    const float* __restrict__ W2, const float* __restrict__ W3,
                                    const float* __restrict__ W4, const float* __restrict__ W5,
                                    const float* __restrict__ W6, _Float16* __restrict__ packed) {
    int idx = blockIdx.x * 256 + threadIdx.x;
    if (idx >= 7 * 4096) return;
    int m = idx >> 12, r = idx & 4095;
    int f = r >> 9, l = (r >> 4) & 31, h = r & 15;
    int kt = f >> 2, nt = f & 3;
    int k = kt * 32 + (h < 8 ? (l >> 4) * 8 + h : 16 + (l >> 4) * 8 + (h - 8));
    int n = nt * 16 + (l & 15);
    const float* W = (m == 0) ? W0 : (m == 1) ? W1 : (m == 2) ? W2 : (m == 3) ? W3
                   : (m == 4) ? W4 : (m == 5) ? W5 : W6;
    packed[idx] = (_Float16)W[k * 64 + n];
}

// Stage A: h = relu(x@W_in+b_in); xl = h@W_lin; a_src = h@W_src; a_dst = h@W_dst
__global__ void node_proj_kernel(const float* __restrict__ xin,
                                 const float* __restrict__ W_in, const float* __restrict__ b_in,
                                 const _Float16* __restrict__ packedW,
                                 float* __restrict__ xh, _Float16* __restrict__ xl,
                                 _Float16* __restrict__ as, _Float16* __restrict__ ad) {
    __shared__ __attribute__((aligned(16))) _Float16 smem[12288 + 8192];
    int  wave = threadIdx.x >> 5, lane = threadIdx.x & 31;
    int  lo = lane & 15, hi = lane >> 4;
    // TDM: W_lin|W_src|W_dst packed block (6144 dwords) -> LDS[0]
    if (threadIdx.x < 32) tdm_stage(0u, packedW, 6144u);
    __syncthreads();
    long wid = (long)blockIdx.x * 8 + wave;
    if (wid >= Nn / 16) return;
    long nb = wid * 16, node = nb + lo;

    float x0 = xin[node * 3], x1 = xin[node * 3 + 1], x2 = xin[node * 3 + 2];
    v16h a0, a1;
    float* xrow = xh + node * 64;
#pragma unroll
    for (int chk = 0; chk < 4; ++chk) {
        int cb = (chk >> 1) * 32 + (chk & 1) * 16 + 8 * hi;
#pragma unroll
        for (int j = 0; j < 8; ++j) {
            int   c = cb + j;
            float h = fmaxf(x0 * W_in[c] + x1 * W_in[64 + c] + x2 * W_in[128 + c] + b_in[c], 0.f);
            xrow[c] = h;
            _Float16 hh = (_Float16)h;
            if (chk == 0) a0[j] = hh; else if (chk == 1) a0[8 + j] = hh;
            else if (chk == 2) a1[j] = hh; else a1[8 + j] = hh;
        }
    }
    _Float16* ldsb = smem + 12288 + wave * 1024;
    v8f d[4];
    gemm64(d, a0, a1, smem,        lane); { Chunks c = transpose_D(ldsb, d, lane); store_chunks(xl, nb, c, lane); }
    gemm64(d, a0, a1, smem + 4096, lane); { Chunks c = transpose_D(ldsb, d, lane); store_chunks(as, nb, c, lane); }
    gemm64(d, a0, a1, smem + 8192, lane); { Chunks c = transpose_D(ldsb, d, lane); store_chunks(ad, nb, c, lane); }
}

// Stage B: delta = posMLP(pos[d]-pos[s]); alpha = attnMLP(a_dst[d]-a_src[s]+delta)
__global__ void edge_alpha_kernel(const int* __restrict__ ei, const float* __restrict__ pos,
                                  const _Float16* __restrict__ as, const _Float16* __restrict__ ad,
                                  const _Float16* __restrict__ packedW,
                                  const float* __restrict__ pW1, const float* __restrict__ pb1,
                                  const float* __restrict__ pg1, const float* __restrict__ pt1,
                                  const float* __restrict__ pb2, const float* __restrict__ pg2,
                                  const float* __restrict__ pt2,
                                  const float* __restrict__ ab1, const float* __restrict__ ag1,
                                  const float* __restrict__ at1,
                                  const float* __restrict__ ab2, const float* __restrict__ ag2,
                                  const float* __restrict__ at2,
                                  _Float16* __restrict__ delta_out, _Float16* __restrict__ alpha_out,
                                  unsigned* __restrict__ amax) {
    __shared__ __attribute__((aligned(16))) _Float16 smem[12288 + 8192];
    int  wave = threadIdx.x >> 5, lane = threadIdx.x & 31;
    int  lo = lane & 15, hi = lane >> 4;
    // TDM: pW2|aW1|aW2 packed block (6144 dwords, matrices 3..5) -> LDS[0]
    if (threadIdx.x < 32) tdm_stage(0u, packedW + 3 * 4096, 6144u);
    __syncthreads();
    long wid = (long)blockIdx.x * 8 + wave;
    if (wid >= (long)ET / 16) return;
    long eb = wid * 16, e = eb + lo;

    int si, di;
    if (e < Ee) { si = ei[e]; di = ei[Ee + e]; } else { si = di = (int)(e - Ee); }
    float dx = pos[(long)di * 3 + 0] - pos[(long)si * 3 + 0];
    float dy = pos[(long)di * 3 + 1] - pos[(long)si * 3 + 1];
    float dz = pos[(long)di * 3 + 2] - pos[(long)si * 3 + 2];

    // pos-MLP layer1 (3->64) in VALU, packed straight into A-fragment layout
    v16h a0, a1;
#pragma unroll
    for (int chk = 0; chk < 4; ++chk) {
        int cb = (chk >> 1) * 32 + (chk & 1) * 16 + 8 * hi;
#pragma unroll
        for (int j = 0; j < 8; ++j) {
            int   c = cb + j;
            float h = dx * pW1[c] + dy * pW1[64 + c] + dz * pW1[128 + c] + pb1[c];
            float s = BN_S * pg1[c];
            h = fmaxf(h * s + pt1[c], 0.f);
            _Float16 hh = (_Float16)h;
            if (chk == 0) a0[j] = hh; else if (chk == 1) a0[8 + j] = hh;
            else if (chk == 2) a1[j] = hh; else a1[8 + j] = hh;
        }
    }
    _Float16* ldsb = smem + 12288 + wave * 1024;
    v8f d[4];

    // pos-MLP layer2 (WMMA) -> delta
    gemm64(d, a0, a1, smem, lane);
    bias_bn_relu(d, pb2, pg2, pt2, lo);
    Chunks dc = transpose_D(ldsb, d, lane);
    store_chunks(delta_out, eb, dc, lane);

    // attn input = a_dst[di] - a_src[si] + delta (f16 chunk gathers)
    const v8h* rs = (const v8h*)(as + (long)si * 64);
    const v8h* rd = (const v8h*)(ad + (long)di * 64);
    v8h q0 = rd[0 + hi] - rs[0 + hi] + dc.c0;
    v8h q1 = rd[2 + hi] - rs[2 + hi] + dc.c1;
    v8h q2 = rd[4 + hi] - rs[4 + hi] + dc.c2;
    v8h q3 = rd[6 + hi] - rs[6 + hi] + dc.c3;
    v16h qa0 = make_afrag(q0, q1), qa1 = make_afrag(q2, q3);

    // attn-MLP layer1 (WMMA)
    gemm64(d, qa0, qa1, smem + 4096, lane);
    bias_bn_relu(d, ab1, ag1, at1, lo);
    Chunks hc = transpose_D(ldsb, d, lane);
    v16h h0 = make_afrag(hc.c0, hc.c1), h1 = make_afrag(hc.c2, hc.c3);

    // attn-MLP layer2 (WMMA) -> alpha (>= 0 after ReLU)
    gemm64(d, h0, h1, smem + 8192, lane);
    bias_bn_relu(d, ab2, ag2, at2, lo);
    Chunks ac = transpose_D(ldsb, d, lane);
    store_chunks(alpha_out, eb, ac, lane);

    // segment max: alpha>=0 so max over zero-extended f16 bit patterns is exact
    unsigned* am = amax + (long)di * 64;
    auto doAtomic = [&](v8h v, int cb) {
#pragma unroll
        for (int j = 0; j < 8; ++j) {
            unsigned short us = __builtin_bit_cast(unsigned short, v[j]);
            atomicMax(&am[cb + j], (unsigned)us);
        }
    };
    doAtomic(ac.c0,  0 + 8 * hi); doAtomic(ac.c1, 16 + 8 * hi);
    doAtomic(ac.c2, 32 + 8 * hi); doAtomic(ac.c3, 48 + 8 * hi);
}

// Stage C: e = exp(alpha - amax[dst]); denom[dst] += e; acc[dst] += e*(xl[src]+delta)
__global__ void edge_accum_kernel(const int* __restrict__ ei, const _Float16* __restrict__ xl,
                                  const _Float16* __restrict__ delta, const _Float16* __restrict__ alpha,
                                  const unsigned* __restrict__ amax,
                                  float* __restrict__ denom, float* __restrict__ acc) {
    long tid = (long)blockIdx.x * blockDim.x + threadIdx.x;
    long e   = tid >> 3;
    int  c0  = (int)(tid & 7) * 8;
    if (e >= ET) return;
    int si, di;
    if (e < Ee) { si = ei[e]; di = ei[Ee + e]; } else { si = di = (int)(e - Ee); }
    v8h al = *(const v8h*)(alpha + e * 64 + c0);
    v8h dl = *(const v8h*)(delta + e * 64 + c0);
    v8h xv = *(const v8h*)(xl + (long)si * 64 + c0);
    const unsigned* am = amax + (long)di * 64 + c0;
    float* dn = denom + (long)di * 64 + c0;
    float* ac = acc + (long)di * 64 + c0;
#pragma unroll
    for (int j = 0; j < 8; ++j) {
        float    a  = (float)al[j];
        _Float16 mh = __builtin_bit_cast(_Float16, (unsigned short)am[j]);
        float    ev = __expf(a - (float)mh);
        atomicAdd(&dn[j], ev);
        atomicAdd(&ac[j], ev * ((float)xv[j] + (float)dl[j]));
    }
}

// Stage D: out = relu((acc/denom)@W_up + b_up) + x; LayerNorm
__global__ void node_out_kernel(const float* __restrict__ accb, const float* __restrict__ den,
                                const float* __restrict__ xh,
                                const _Float16* __restrict__ packedW,
                                const float* __restrict__ b_up, const float* __restrict__ lng,
                                const float* __restrict__ lnb, float* __restrict__ out) {
    __shared__ __attribute__((aligned(16))) _Float16 smem[4096];
    int  wave = threadIdx.x >> 5, lane = threadIdx.x & 31;
    int  lo = lane & 15, hi = lane >> 4;
    // TDM: W_up packed (2048 dwords, matrix 6) -> LDS[0]
    if (threadIdx.x < 32) tdm_stage(0u, packedW + 6 * 4096, 2048u);
    __syncthreads();
    long wid = (long)blockIdx.x * 8 + wave;
    if (wid >= Nn / 16) return;
    long nb = wid * 16, node = nb + lo;

    const float* ar = accb + node * 64;
    const float* dr = den + node * 64;
    v16h a0, a1;
#pragma unroll
    for (int chk = 0; chk < 4; ++chk) {
        int cb = (chk >> 1) * 32 + (chk & 1) * 16 + 8 * hi;
#pragma unroll
        for (int j = 0; j < 8; ++j) {
            _Float16 hh = (_Float16)(ar[cb + j] / dr[cb + j]);
            if (chk == 0) a0[j] = hh; else if (chk == 1) a0[8 + j] = hh;
            else if (chk == 2) a1[j] = hh; else a1[8 + j] = hh;
        }
    }
    v8f d[4];
    gemm64(d, a0, a1, smem, lane);
#pragma unroll
    for (int nt = 0; nt < 4; ++nt) {
        int   c  = nt * 16 + lo;
        float bu = b_up[c];
#pragma unroll
        for (int v = 0; v < 8; ++v) {
            long m = nb + 8 * hi + v;
            d[nt][v] = fmaxf(d[nt][v] + bu, 0.f) + xh[m * 64 + c];
        }
    }
    // LayerNorm: row m = nb + 8*hi + v lives entirely within one 16-lane half
#pragma unroll
    for (int v = 0; v < 8; ++v) {
        float s  = d[0][v] + d[1][v] + d[2][v] + d[3][v];
        float s2 = d[0][v] * d[0][v] + d[1][v] * d[1][v] + d[2][v] * d[2][v] + d[3][v] * d[3][v];
#pragma unroll
        for (int m = 1; m < 16; m <<= 1) {
            s  += __shfl_xor(s, m, 32);
            s2 += __shfl_xor(s2, m, 32);
        }
        float mean = s * (1.f / 64.f);
        float var  = s2 * (1.f / 64.f) - mean * mean;
        float inv  = rsqrtf(var + 1e-5f);
        long  row  = nb + 8 * hi + v;
#pragma unroll
        for (int nt = 0; nt < 4; ++nt) {
            int c = nt * 16 + lo;
            out[row * 64 + c] = (d[nt][v] - mean) * inv * lng[c] + lnb[c];
        }
    }
}

// ----------------------------------------------------------------- launch ---

extern "C" void kernel_launch(void* const* d_in, const int* in_sizes, int n_in,
                              void* d_out, int out_size, void* d_ws, size_t ws_size,
                              hipStream_t stream) {
    const float* xin  = (const float*)d_in[0];
    const float* pos  = (const float*)d_in[1];
    const int*   ei   = (const int*)d_in[2];
    const float* W_in = (const float*)d_in[3];  const float* b_in = (const float*)d_in[4];
    const float* W_lin= (const float*)d_in[5];
    const float* W_src= (const float*)d_in[6];  const float* W_dst= (const float*)d_in[7];
    const float* pW1  = (const float*)d_in[8];  const float* pb1 = (const float*)d_in[9];
    const float* pg1  = (const float*)d_in[10]; const float* pt1 = (const float*)d_in[11];
    const float* pW2  = (const float*)d_in[12]; const float* pb2 = (const float*)d_in[13];
    const float* pg2  = (const float*)d_in[14]; const float* pt2 = (const float*)d_in[15];
    const float* aW1  = (const float*)d_in[16]; const float* ab1 = (const float*)d_in[17];
    const float* ag1  = (const float*)d_in[18]; const float* at1 = (const float*)d_in[19];
    const float* aW2  = (const float*)d_in[20]; const float* ab2 = (const float*)d_in[21];
    const float* ag2  = (const float*)d_in[22]; const float* at2 = (const float*)d_in[23];
    const float* W_up = (const float*)d_in[24]; const float* b_up= (const float*)d_in[25];
    const float* ln_g = (const float*)d_in[26]; const float* ln_b= (const float*)d_in[27];
    float* outp = (float*)d_out;

    char*  ws = (char*)d_ws;
    size_t o  = 0;
    float*     xh    = (float*)(ws + o);     o += (size_t)Nn * 64 * 4;
    _Float16*  xl    = (_Float16*)(ws + o);  o += (size_t)Nn * 64 * 2;
    _Float16*  asrc  = (_Float16*)(ws + o);  o += (size_t)Nn * 64 * 2;
    _Float16*  adst  = (_Float16*)(ws + o);  o += (size_t)Nn * 64 * 2;
    _Float16*  delta = (_Float16*)(ws + o);  o += (size_t)ET * 64 * 2;
    _Float16*  alpha = (_Float16*)(ws + o);  o += (size_t)ET * 64 * 2;
    unsigned*  amax  = (unsigned*)(ws + o);  o += (size_t)Nn * 64 * 4;
    float*     denom = (float*)(ws + o);     o += (size_t)Nn * 64 * 4;
    float*     acc   = (float*)(ws + o);     o += (size_t)Nn * 64 * 4;
    _Float16*  packedW = (_Float16*)(ws + o); o += (size_t)7 * 4096 * 2;

    // pre-pack weights into f16 B-fragment layout (TDM source block)
    pack_weights_kernel<<<(7 * 4096 + 255) / 256, 256, 0, stream>>>(
        W_lin, W_src, W_dst, pW2, aW1, aW2, W_up, packedW);

    // zero amax+denom+acc (contiguous 3*N*64 words) every call
    long zn = 3L * Nn * 64;
    zero_kernel<<<(int)((zn + 255) / 256), 256, 0, stream>>>(amax, zn);

    int nodeBlocks = (Nn / 16 + 7) / 8;        // 8 waves/block, 16 nodes/wave
    node_proj_kernel<<<nodeBlocks, 256, 0, stream>>>(xin, W_in, b_in, packedW,
                                                     xh, xl, asrc, adst);

    int edgeBlocks = (ET / 16 + 7) / 8;        // 8 waves/block, 16 edges/wave
    edge_alpha_kernel<<<edgeBlocks, 256, 0, stream>>>(ei, pos, asrc, adst, packedW,
                                                      pW1, pb1, pg1, pt1, pb2, pg2, pt2,
                                                      ab1, ag1, at1, ab2, ag2, at2,
                                                      delta, alpha, amax);

    long accThreads = (long)ET * 8;            // 8 threads per edge (8 channels each)
    edge_accum_kernel<<<(int)((accThreads + 255) / 256), 256, 0, stream>>>(ei, xl, delta, alpha,
                                                                           amax, denom, acc);

    node_out_kernel<<<nodeBlocks, 256, 0, stream>>>(acc, denom, xh, packedW,
                                                    b_up, ln_g, ln_b, outp);
}
